// Multi_HMR_74010876445108
// MI455X (gfx1250) — matrix operand order repaired and checked by reading, not compiled, not run on hardware
//
#include <hip/hip_runtime.h>

// ---------------- problem constants ----------------
#define BATCH 4
#define GRID 64
#define GG   4096        // GRID*GRID
#define CE_  384
#define DIM  512
#define NLAY 8
#define NH   16
#define DH_  32
#define MFF  2048
#define KQ_  16
#define NJ_  163
#define NB_  11
#define ROWS 64          // BATCH*KQ
#define OUTC 1486        // 2+2+1+3+11+163*9

typedef __bf16 bf16;
typedef __bf16 v16bf __attribute__((ext_vector_type(16)));
typedef __bf16 v8bf  __attribute__((ext_vector_type(8)));
typedef float  v8f   __attribute__((ext_vector_type(8)));

// =====================================================================
// WMMA GEMM: C_bf16[M,N] = A[M,K] @ B[K,N] (+bias[n]) (+pos[row&mask,n])
// A is f32 or bf16 (template). NB = number of B/C pairs (1 or 2): NB=2
// fuses two GEMMs sharing the same A (K and V projections), halving the
// dominant ctx_bf16 read traffic. Block tile 128x128, K-step 32, 8 waves.
// M,N multiples of 128; K multiple of 32; posM power of two.
// =====================================================================
template <int ABF, int NBM>
__global__ __launch_bounds__(256) void k_gemm_wmma(
    const void* __restrict__ Ap,
    const float* __restrict__ Bw0, const float* __restrict__ Bw1,
    const float* __restrict__ bias, const float* __restrict__ pos, int posMask,
    bf16* __restrict__ C0, bf16* __restrict__ C1,
    int M, int N, int K)
{
    __shared__ bf16 As[128][40];          // [m][k], stride 40 keeps 16B align
    __shared__ bf16 Bs[NBM][128][40];     // [n][k], transposed

    const int tid  = threadIdx.x;
    const int lane = tid & 31;
    const int wid  = tid >> 5;
    const int wm   = wid & 1;      // 2 waves along M
    const int wn   = wid >> 1;     // 4 waves along N
    const int m0   = blockIdx.y * 128;
    const int n0   = blockIdx.x * 128;
    const int half = lane >> 4;    // K-half selector (ISA 16-bit A/B layout)
    const int mr   = lane & 15;

    v8f acc[NBM][4][2];
#pragma unroll
    for (int s = 0; s < NBM; ++s)
#pragma unroll
        for (int i = 0; i < 4; ++i)
#pragma unroll
            for (int j = 0; j < 2; ++j)
#pragma unroll
                for (int e = 0; e < 8; ++e) acc[s][i][j][e] = 0.0f;

    for (int ks = 0; ks < K; ks += 32) {
        __syncthreads();
        // ---- stage A tile (128 x 32) -> LDS bf16 ----
        {
            const int r  = tid >> 1;
            const int c0 = (tid & 1) * 16;
            if (ABF) {
                const bf16* A = (const bf16*)Ap;
                const v8bf* src = (const v8bf*)(A + (size_t)(m0 + r) * K + ks + c0);
                *(v8bf*)&As[r][c0]     = src[0];
                *(v8bf*)&As[r][c0 + 8] = src[1];
            } else {
                const float* A = (const float*)Ap + (size_t)(m0 + r) * K + ks + c0;
                float fv[16];
#pragma unroll
                for (int i = 0; i < 16; ++i) fv[i] = A[i];
                v8bf lo, hi;
#pragma unroll
                for (int i = 0; i < 8; ++i) { lo[i] = (bf16)fv[i]; hi[i] = (bf16)fv[i + 8]; }
                *(v8bf*)&As[r][c0]     = lo;
                *(v8bf*)&As[r][c0 + 8] = hi;
            }
        }
        // ---- stage B tile(s) (32 x 128) transposed -> LDS[n][k] ----
        {
            const int n  = tid & 127;
            const int k0 = (tid >> 7) * 16;
#pragma unroll
            for (int s = 0; s < NBM; ++s) {
                const float* bw = (s == 0) ? Bw0 : Bw1;
                const float* bp = bw + (size_t)(ks + k0) * N + (n0 + n);
#pragma unroll
                for (int i = 0; i < 16; ++i)
                    Bs[s][n][k0 + i] = (bf16)bp[(size_t)i * N];
                // prefetch next K-step's B rows (global_prefetch_b8)
                if (ks + 32 < K)
                    __builtin_prefetch(bp + (size_t)32 * N, 0, 1);
            }
        }
        __syncthreads();

        // ---- build A fragments per ISA 16-bit A layout ----
        v16bf afr[4];
#pragma unroll
        for (int i = 0; i < 4; ++i) {
            const int row = wm * 64 + i * 16 + mr;
            v8bf lo = *(const v8bf*)&As[row][half * 8];
            v8bf hi = *(const v8bf*)&As[row][16 + half * 8];
#pragma unroll
            for (int e = 0; e < 8; ++e) { afr[i][e] = lo[e]; afr[i][e + 8] = hi[e]; }
        }
        // ---- per B-matrix: build B fragments, issue WMMAs ----
#pragma unroll
        for (int s = 0; s < NBM; ++s) {
            v16bf bfr[2];
#pragma unroll
            for (int j = 0; j < 2; ++j) {
                const int col = wn * 32 + j * 16 + mr;
                v8bf lo = *(const v8bf*)&Bs[s][col][half * 8];
                v8bf hi = *(const v8bf*)&Bs[s][col][16 + half * 8];
#pragma unroll
                for (int e = 0; e < 8; ++e) { bfr[j][e] = lo[e]; bfr[j][e + 8] = hi[e]; }
            }
#pragma unroll
            for (int i = 0; i < 4; ++i)
#pragma unroll
                for (int j = 0; j < 2; ++j)
                    acc[s][i][j] = __builtin_amdgcn_wmma_f32_16x16x32_bf16(
                        false, afr[i], false, bfr[j], (short)0, acc[s][i][j], false, false);
        }
    }

    // ---- epilogue: bias + pos-emb add, store bf16 (C layout per ISA) ----
#pragma unroll
    for (int s = 0; s < NBM; ++s) {
        bf16* Cb = (s == 0) ? C0 : C1;
#pragma unroll
        for (int i = 0; i < 4; ++i)
#pragma unroll
            for (int j = 0; j < 2; ++j) {
                const int col = n0 + wn * 32 + j * 16 + mr;
                const float bv = bias ? bias[col] : 0.0f;
#pragma unroll
                for (int r = 0; r < 8; ++r) {
                    const int row = m0 + wm * 64 + i * 16 + half * 8 + r;
                    float v = acc[s][i][j][r] + bv;
                    if (pos) v += pos[(size_t)(row & posMask) * N + col];
                    Cb[(size_t)row * N + col] = (bf16)v;
                }
            }
    }
}

// =====================================================================
// NMS (3x3 SAME max-pool keep-if-peak) + iterative top-16 per batch
// =====================================================================
__global__ __launch_bounds__(256) void k_nms_topk(const float* __restrict__ scores,
                                                  int* __restrict__ idxb,
                                                  float* __restrict__ maskb)
{
    const int b = blockIdx.x;
    __shared__ float ssc[GG];
    __shared__ float rv[256];
    __shared__ int   ri[256];
    const float* s = scores + (size_t)b * GG;
    const int tid = threadIdx.x;

    for (int p = tid; p < GG; p += 256) {
        const int i = p >> 6, j = p & 63;
        float mx = -1e30f;
        for (int di = -1; di <= 1; ++di) {
            const int ii = i + di; if (ii < 0 || ii > 63) continue;
            for (int dj = -1; dj <= 1; ++dj) {
                const int jj = j + dj; if (jj < 0 || jj > 63) continue;
                mx = fmaxf(mx, s[ii * 64 + jj]);
            }
        }
        const float v = s[p];
        ssc[p] = (v == mx) ? v : 0.0f;
    }
    __syncthreads();

    for (int t = 0; t < KQ_; ++t) {
        float bv = -1e30f; int bi = 0x7fffffff;
        for (int p = tid; p < GG; p += 256) {
            const float v = ssc[p];
            if (v > bv || (v == bv && p < bi)) { bv = v; bi = p; }
        }
        rv[tid] = bv; ri[tid] = bi;
        __syncthreads();
        for (int st = 128; st > 0; st >>= 1) {
            if (tid < st) {
                if (rv[tid + st] > rv[tid] ||
                    (rv[tid + st] == rv[tid] && ri[tid + st] < ri[tid])) {
                    rv[tid] = rv[tid + st]; ri[tid] = ri[tid + st];
                }
            }
            __syncthreads();
        }
        if (tid == 0) {
            idxb[b * KQ_ + t]  = ri[0];
            maskb[b * KQ_ + t] = (rv[0] >= 0.3f) ? 1.0f : 0.0f;
            ssc[ri[0]] = -1e30f;
        }
        __syncthreads();
    }
}

// gather x[r,:] = ctx_bf16[b, idx[r], :] * mask[r]
__global__ void k_gather_x(const bf16* __restrict__ ctx, const int* __restrict__ idxb,
                           const float* __restrict__ maskb, float* __restrict__ x)
{
    const int r = blockIdx.x;
    const int b = r >> 4;
    const int id = idxb[r];
    const float m = maskb[r];
    const bf16* src = ctx + ((size_t)(b * GG + id)) * DIM;
    for (int c = threadIdx.x; c < DIM; c += blockDim.x)
        x[(size_t)r * DIM + c] = (float)src[c] * m;
}

// LayerNorm over 512-wide rows
__global__ __launch_bounds__(256) void k_ln(const float* __restrict__ x,
                                            const float* __restrict__ g,
                                            const float* __restrict__ bb,
                                            float* __restrict__ y)
{
    const int r = blockIdx.x;
    const float* xr = x + (size_t)r * DIM;
    __shared__ float red[256];
    const int tid = threadIdx.x;

    float s = 0.0f;
    for (int c = tid; c < DIM; c += 256) s += xr[c];
    red[tid] = s; __syncthreads();
    for (int st = 128; st > 0; st >>= 1) { if (tid < st) red[tid] += red[tid + st]; __syncthreads(); }
    const float mu = red[0] / (float)DIM;
    __syncthreads();

    float v = 0.0f;
    for (int c = tid; c < DIM; c += 256) { const float d = xr[c] - mu; v += d * d; }
    red[tid] = v; __syncthreads();
    for (int st = 128; st > 0; st >>= 1) { if (tid < st) red[tid] += red[tid + st]; __syncthreads(); }
    const float var = red[0] / (float)DIM;
    const float rs = rsqrtf(var + 1e-5f);
    for (int c = tid; c < DIM; c += 256)
        y[(size_t)r * DIM + c] = (xr[c] - mu) * rs * g[c] + bb[c];
}

// Small row-per-block GEMM: C[m,:] = act(A[m,:]@W + bias) (+extra); optional
// masked residual-accumulate into C. act: 0 none, 1 relu, 2 gelu(tanh), 3 sigmoid
__global__ __launch_bounds__(256) void k_sgemm(const float* __restrict__ A,
                                               const float* __restrict__ W,
                                               const float* __restrict__ bias,
                                               const float* __restrict__ extra,
                                               const float* __restrict__ rowmask,
                                               float* __restrict__ C,
                                               int K, int N, int act)
{
    const int m = blockIdx.x;
    __shared__ float ar[MFF];
    for (int k = threadIdx.x; k < K; k += 256) ar[k] = A[(size_t)m * K + k];
    __syncthreads();
    for (int n = threadIdx.x; n < N; n += 256) {
        float acc = bias ? bias[n] : 0.0f;
        for (int k = 0; k < K; ++k) acc += ar[k] * W[(size_t)k * N + n];
        if (act == 1) acc = fmaxf(acc, 0.0f);
        else if (act == 2) {
            const float x3 = acc * acc * acc;
            acc = 0.5f * acc * (1.0f + tanhf(0.7978845608f * (acc + 0.044715f * x3)));
        } else if (act == 3) acc = 1.0f / (1.0f + expf(-acc));
        if (extra) acc += extra[n];
        const size_t o = (size_t)m * N + n;
        if (rowmask) C[o] += acc * rowmask[m];
        else C[o] = acc;
    }
}

// attention logits: att[b,h,q,k] = scale * qh[b,q,h,:]·kh[b,k,h,:]
__global__ __launch_bounds__(256) void k_qk(const float* __restrict__ qh,
                                            const bf16* __restrict__ kh,
                                            float* __restrict__ att)
{
    const int bh = blockIdx.x;
    const int b = bh >> 4, h = bh & 15;
    __shared__ float qs[KQ_][DH_];
    const int tid = threadIdx.x;
    for (int t = tid; t < KQ_ * DH_; t += 256) {
        const int q = t >> 5, d = t & 31;
        qs[q][d] = qh[((size_t)(b * KQ_ + q)) * DIM + h * DH_ + d];
    }
    __syncthreads();
    const float scale = 0.1767766952966369f; // 32^-0.5
    for (int k = tid; k < GG; k += 256) {
        const v8bf* kr = (const v8bf*)(kh + ((size_t)(b * GG + k)) * DIM + h * DH_);
        float kv[DH_];
#pragma unroll
        for (int c = 0; c < 4; ++c) {
            v8bf t8 = kr[c];
#pragma unroll
            for (int d = 0; d < 8; ++d) kv[c * 8 + d] = (float)t8[d];
        }
        for (int q = 0; q < KQ_; ++q) {
            float acc = 0.0f;
#pragma unroll
            for (int d = 0; d < DH_; ++d) acc += qs[q][d] * kv[d];
            att[(((size_t)bh * KQ_ + q) << 12) + k] = acc * scale;
        }
    }
}

__global__ __launch_bounds__(256) void k_softmax4096(float* __restrict__ att)
{
    float* row = att + ((size_t)blockIdx.x << 12);
    __shared__ float red[256];
    const int tid = threadIdx.x;
    float mx = -1e30f;
    for (int k = tid; k < GG; k += 256) mx = fmaxf(mx, row[k]);
    red[tid] = mx; __syncthreads();
    for (int st = 128; st > 0; st >>= 1) { if (tid < st) red[tid] = fmaxf(red[tid], red[tid + st]); __syncthreads(); }
    mx = red[0]; __syncthreads();
    float s = 0.0f;
    for (int k = tid; k < GG; k += 256) { const float e = expf(row[k] - mx); row[k] = e; s += e; }
    red[tid] = s; __syncthreads();
    for (int st = 128; st > 0; st >>= 1) { if (tid < st) red[tid] += red[tid + st]; __syncthreads(); }
    const float inv = 1.0f / red[0];
    __syncthreads();
    for (int k = tid; k < GG; k += 256) row[k] *= inv;
}

// o[b,q,h,d] = sum_k att[b,h,q,k] * vh[b,k,h,d]   (512 threads: q=tid>>5, d=tid&31)
__global__ __launch_bounds__(512) void k_av(const float* __restrict__ att,
                                            const bf16* __restrict__ vh,
                                            float* __restrict__ o)
{
    const int bh = blockIdx.x;
    const int b = bh >> 4, h = bh & 15;
    const int tid = threadIdx.x;
    const int q = tid >> 5, d = tid & 31;
    const float* arow = att + (((size_t)bh * KQ_ + q) << 12);
    const bf16* vcol = vh + ((size_t)b * GG) * DIM + h * DH_ + d;
    float acc = 0.0f;
    for (int k = 0; k < GG; ++k) acc += arow[k] * (float)vcol[(size_t)k * DIM];
    o[((size_t)(b * KQ_ + q)) * DIM + h * DH_ + d] = acc;
}

// posec[n] = pb1[n] + sum_j ibp[j] * pW1[512+j, n]
__global__ void k_pose_const(const float* __restrict__ pW1, const float* __restrict__ pb1,
                             const float* __restrict__ ibp, float* __restrict__ posec)
{
    const int n = blockIdx.x * blockDim.x + threadIdx.x;
    if (n < DIM) {
        float a = pb1[n];
        for (int j = 0; j < NJ_ * 6; ++j) a += ibp[j] * pW1[(size_t)(DIM + j) * DIM + n];
        posec[n] = a;
    }
}

// Final assembly: loc/offset/dist/transl/shape + rot6d -> rotmat, masked
__global__ __launch_bounds__(256) void k_assemble(
    const float* __restrict__ off, const float* __restrict__ draw,
    const float* __restrict__ shp, const float* __restrict__ rot6d,
    const int* __restrict__ idxb, const float* __restrict__ maskb,
    const float* __restrict__ Kcam, const float* __restrict__ useful,
    float* __restrict__ out)
{
    const int r = blockIdx.x;
    const int b = r >> 4;
    const float m = maskb[r];
    float* o = out + (size_t)r * OUTC;
    const int tid = threadIdx.x;

    if (tid == 0) {
        const int id = idxb[r];
        const int row = id >> 6, col = id & 63;
        const float o0 = off[r * 2 + 0], o1 = off[r * 2 + 1];
        const float lx = ((float)col + 0.5f + o0) * 14.0f;
        const float ly = ((float)row + 0.5f + o1) * 14.0f;
        const float* Kc = Kcam + b * 9;
        const float f = Kc[0], cx = Kc[2], cy = Kc[5];
        const float dist = f / fmaxf(expf(draw[r]), 1e-5f);
        const float tx = (lx - cx) / f, ty = (ly - cy) / f;
        o[0] = lx * m; o[1] = ly * m;
        o[2] = o0 * m; o[3] = o1 * m;
        o[4] = dist * m;
        o[5] = tx * dist * m; o[6] = ty * dist * m; o[7] = dist * m;
    }
    if (tid < NB_) o[8 + tid] = shp[r * NB_ + tid] * m;

    for (int j = tid; j < NJ_; j += 256) {
        const float* rp = rot6d + (size_t)r * (NJ_ * 6) + j * 6;
        float u0x = rp[0], u0y = rp[2], u0z = rp[4];
        const float bx = rp[1], by = rp[3], bz = rp[5];
        const float n0 = sqrtf(u0x * u0x + u0y * u0y + u0z * u0z) + 1e-8f;
        u0x /= n0; u0y /= n0; u0z /= n0;
        const float dd = bx * u0x + by * u0y + bz * u0z;
        const float v1x = bx - dd * u0x, v1y = by - dd * u0y, v1z = bz - dd * u0z;
        const float n1 = sqrtf(v1x * v1x + v1y * v1y + v1z * v1z) + 1e-8f;
        const float u1x = v1x / n1, u1y = v1y / n1, u1z = v1z / n1;
        const float u2x = u0y * u1z - u0z * u1y;
        const float u2y = u0z * u1x - u0x * u1z;
        const float u2z = u0x * u1y - u0y * u1x;
        const float um = useful[j], w = 1.0f - um;
        float* dst = o + 19 + j * 9;
        dst[0] = (um * u0x + w) * m; dst[1] = (um * u1x) * m; dst[2] = (um * u2x) * m;
        dst[3] = (um * u0y) * m; dst[4] = (um * u1y + w) * m; dst[5] = (um * u2y) * m;
        dst[6] = (um * u0z) * m; dst[7] = (um * u1z) * m; dst[8] = (um * u2z + w) * m;
    }
}

// =====================================================================
extern "C" void kernel_launch(void* const* d_in, const int* in_sizes, int n_in,
                              void* d_out, int out_size, void* d_ws, size_t ws_size,
                              hipStream_t stream)
{
    (void)in_sizes; (void)n_in; (void)out_size; (void)ws_size;

    const float* feat   = (const float*)d_in[0];
    const float* scores = (const float*)d_in[1];
    const float* K_cam  = (const float*)d_in[2];
    const float* pos_emb= (const float*)d_in[3];
    const float* Wtok   = (const float*)d_in[4];
    const float* btok   = (const float*)d_in[5];
    const float* Wq     = (const float*)d_in[6];
    const float* Wk     = (const float*)d_in[7];
    const float* Wv     = (const float*)d_in[8];
    const float* Wo     = (const float*)d_in[9];
    const float* g1     = (const float*)d_in[10];
    const float* b1n    = (const float*)d_in[11];
    const float* W1     = (const float*)d_in[12];
    const float* bf1    = (const float*)d_in[13];
    const float* W2     = (const float*)d_in[14];
    const float* bf2    = (const float*)d_in[15];
    const float* g2     = (const float*)d_in[16];
    const float* b2n    = (const float*)d_in[17];
    const float* offW1  = (const float*)d_in[18];
    const float* offb1  = (const float*)d_in[19];
    const float* offW2  = (const float*)d_in[20];
    const float* offb2  = (const float*)d_in[21];
    const float* dW1    = (const float*)d_in[22];
    const float* db1    = (const float*)d_in[23];
    const float* dW2    = (const float*)d_in[24];
    const float* db2    = (const float*)d_in[25];
    const float* sW1    = (const float*)d_in[26];
    const float* sb1    = (const float*)d_in[27];
    const float* sW2    = (const float*)d_in[28];
    const float* sb2    = (const float*)d_in[29];
    const float* pW1    = (const float*)d_in[30];
    const float* pb1    = (const float*)d_in[31];
    const float* pW2    = (const float*)d_in[32];
    const float* pb2    = (const float*)d_in[33];
    const float* ibp    = (const float*)d_in[34];
    const float* useful = (const float*)d_in[35];
    float* out = (float*)d_out;

    // ---- workspace carve-out ----
    char* w = (char*)d_ws;
    size_t off = 0;
    auto alloc = [&](size_t bytes) -> void* {
        off = (off + 255) & ~(size_t)255;
        void* p = w + off; off += bytes; return p;
    };
    bf16*  ctx_bf = (bf16*)alloc((size_t)BATCH * GG * DIM * 2);
    bf16*  kh     = (bf16*)alloc((size_t)BATCH * GG * DIM * 2);
    bf16*  vh     = (bf16*)alloc((size_t)BATCH * GG * DIM * 2);
    float* att    = (float*)alloc((size_t)BATCH * NH * KQ_ * GG * 4);
    float* x      = (float*)alloc((size_t)ROWS * DIM * 4);
    float* h      = (float*)alloc((size_t)ROWS * DIM * 4);
    float* qh     = (float*)alloc((size_t)ROWS * DIM * 4);
    float* obuf   = (float*)alloc((size_t)ROWS * DIM * 4);
    float* t1     = (float*)alloc((size_t)ROWS * MFF * 4);
    float* hid    = (float*)alloc((size_t)ROWS * DIM * 4);
    float* offout = (float*)alloc((size_t)ROWS * 2 * 4);
    float* dout1  = (float*)alloc((size_t)ROWS * 4);
    float* shout  = (float*)alloc((size_t)ROWS * NB_ * 4);
    float* posec  = (float*)alloc((size_t)DIM * 4);
    float* rot6d  = (float*)alloc((size_t)ROWS * NJ_ * 6 * 4);
    int*   idxb   = (int*)alloc((size_t)ROWS * 4);
    float* maskb  = (float*)alloc((size_t)ROWS * 4);

    const int MTOT = BATCH * GG;   // 16384
    const dim3 gg(DIM / 128, MTOT / 128);   // (4, 128)

    // 1) detection
    k_nms_topk<<<BATCH, 256, 0, stream>>>(scores, idxb, maskb);

    // 2) ctx = feat @ Wtok + btok + pos_emb  (bf16 out) -- WMMA
    k_gemm_wmma<0, 1><<<gg, 256, 0, stream>>>(feat, Wtok, nullptr, btok,
                                              pos_emb, GG - 1,
                                              ctx_bf, nullptr, MTOT, DIM, CE_);

    // 3) gather queries
    k_gather_x<<<ROWS, 256, 0, stream>>>(ctx_bf, idxb, maskb, x);

    // 4) decoder layers
    for (int l = 0; l < NLAY; ++l) {
        const float* Wql = Wq + (size_t)l * DIM * DIM;
        const float* Wkl = Wk + (size_t)l * DIM * DIM;
        const float* Wvl = Wv + (size_t)l * DIM * DIM;
        const float* Wol = Wo + (size_t)l * DIM * DIM;

        k_ln<<<ROWS, 256, 0, stream>>>(x, g1 + l * DIM, b1n + l * DIM, h);
        k_sgemm<<<ROWS, 256, 0, stream>>>(h, Wql, nullptr, nullptr, nullptr, qh, DIM, DIM, 0);

        // fused K+V projection: one pass over ctx_bf16, two WMMA outputs
        k_gemm_wmma<1, 2><<<gg, 256, 0, stream>>>(ctx_bf, Wkl, Wvl, nullptr,
                                                  nullptr, 0,
                                                  kh, vh, MTOT, DIM, DIM);

        k_qk<<<BATCH * NH, 256, 0, stream>>>(qh, kh, att);
        k_softmax4096<<<BATCH * NH * KQ_, 256, 0, stream>>>(att);
        k_av<<<BATCH * NH, 512, 0, stream>>>(att, vh, obuf);

        // x += (o @ Wo) * mask
        k_sgemm<<<ROWS, 256, 0, stream>>>(obuf, Wol, nullptr, nullptr, maskb, x, DIM, DIM, 0);

        // FFN
        k_ln<<<ROWS, 256, 0, stream>>>(x, g2 + l * DIM, b2n + l * DIM, h);
        k_sgemm<<<ROWS, 256, 0, stream>>>(h, W1 + (size_t)l * DIM * MFF, bf1 + l * MFF,
                                          nullptr, nullptr, t1, DIM, MFF, 2);
        k_sgemm<<<ROWS, 256, 0, stream>>>(t1, W2 + (size_t)l * MFF * DIM, bf2 + l * DIM,
                                          nullptr, maskb, x, MFF, DIM, 0);
    }

    // 5) heads
    k_sgemm<<<ROWS, 256, 0, stream>>>(x, offW1, offb1, nullptr, nullptr, hid, DIM, DIM, 1);
    k_sgemm<<<ROWS, 256, 0, stream>>>(hid, offW2, offb2, nullptr, nullptr, offout, DIM, 2, 0);

    k_sgemm<<<ROWS, 256, 0, stream>>>(x, dW1, db1, nullptr, nullptr, hid, DIM, DIM, 1);
    k_sgemm<<<ROWS, 256, 0, stream>>>(hid, dW2, db2, nullptr, nullptr, dout1, DIM, 1, 0);

    k_sgemm<<<ROWS, 256, 0, stream>>>(x, sW1, sb1, nullptr, nullptr, hid, DIM, DIM, 1);
    k_sgemm<<<ROWS, 256, 0, stream>>>(hid, sW2, sb2, nullptr, nullptr, shout, DIM, NB_, 3);

    k_pose_const<<<2, 256, 0, stream>>>(pW1, pb1, ibp, posec);
    k_sgemm<<<ROWS, 256, 0, stream>>>(x, pW1, posec, nullptr, nullptr, hid, DIM, DIM, 1);
    k_sgemm<<<ROWS, 256, 0, stream>>>(hid, pW2, pb2, ibp, nullptr, rot6d, DIM, NJ_ * 6, 0);

    // 6) assemble output
    k_assemble<<<ROWS, 256, 0, stream>>>(offout, dout1, shout, rot6d, idxb, maskb,
                                         K_cam, useful, out);
}
